// GeoMix2_33440615367378
// MI455X (gfx1250) — matrix operand order repaired and checked by reading, not compiled, not run on hardware
//
#include <hip/hip_runtime.h>
#include <hip/hip_fp16.h>

// ---------------------------------------------------------------------------
// GCN 2-layer pipeline for MI455X (gfx1250), wave32 + WMMA.
//   h = agg( x@W1+b1 ); h = relu(BN(h)); out = agg( h@W2+b2 )
// GEMMs run on v_wmma_f32_16x16x32_f16 (f16 inputs, f32 accumulate, K=512).
// Each wave computes a 16x64 output strip (4 accumulators / 1 shared A frag).
// Double-buffered fragment loads are pinned with sched_barrier(0) so the
// backend cannot serialize load->wmma and must overlap them.
// ---------------------------------------------------------------------------

typedef _Float16 h8  __attribute__((ext_vector_type(8)));
typedef _Float16 h16 __attribute__((ext_vector_type(16)));
typedef float    f8  __attribute__((ext_vector_type(8)));

static constexpr int C = 512;          // IN_C == HID == OUT_C == 512
static constexpr float BN_EPS = 1e-5f;

union Frag { h16 v; h8 h[2]; };

// A 16x32 f16 fragment (ISA 7.12.2): lane m=lane&15;
//   lanes<16 hold K {kk..kk+7, kk+16..kk+23}, lanes>=16 {kk+8..kk+15, kk+24..kk+31}
__device__ __forceinline__ void load_a(Frag& f, const _Float16* __restrict__ aRow,
                                       int kk, int grp) {
    f.h[0] = *(const h8*)(aRow + kk + grp * 8);
    f.h[1] = *(const h8*)(aRow + kk + 16 + grp * 8);
}

// B 32x16 f16 fragment from transposed weights: lane n=lane&15;
//   lanes<16 hold K {kk..kk+7, kk+16..kk+23}, lanes>=16 {kk+8..kk+15, kk+24..kk+31}
__device__ __forceinline__ void load_b(Frag& f, const _Float16* __restrict__ bRow,
                                       int kk, int grp) {
    f.h[0] = *(const h8*)(bRow + kk + grp * 8);
    f.h[1] = *(const h8*)(bRow + kk + 16 + grp * 8);
}

// ---------------- degree / normalization --------------------------------
__global__ void k_zero_f32(float* __restrict__ p, int n) {
    int i = blockIdx.x * blockDim.x + threadIdx.x;
    if (i < n) p[i] = 0.0f;
}

__global__ void k_deg_count(const int* __restrict__ dst, float* __restrict__ deg, int E) {
    int e = blockIdx.x * blockDim.x + threadIdx.x;
    if (e < E) atomicAdd(&deg[dst[e]], 1.0f);
}

// deg -> dinv in place (self loop adds 1, so deg >= 1 always)
__global__ void k_dinv(float* __restrict__ deg, int n) {
    int i = blockIdx.x * blockDim.x + threadIdx.x;
    if (i < n) deg[i] = rsqrtf(deg[i] + 1.0f);
}

// ---------------- precision conversion ----------------------------------
__global__ void k_cvt_f16(const float* __restrict__ in, _Float16* __restrict__ out, int n) {
    int i = blockIdx.x * blockDim.x + threadIdx.x;
    if (i < n) out[i] = (_Float16)in[i];
}

// W [k][n] row-major f32 -> Wt [n][k] f16  (so B fragments load contiguously)
__global__ void k_cvt_transpose(const float* __restrict__ W, _Float16* __restrict__ Wt) {
    int n = blockIdx.x;                       // 512 blocks
    for (int k = threadIdx.x; k < C; k += blockDim.x)
        Wt[(size_t)n * C + k] = (_Float16)W[(size_t)k * C + n];
}

// ---------------- WMMA GEMM:  Cout[M,512] = A[M,512] * B[512,512] + bias --
// A row-major f16, Bt = B^T row-major f16 ([n][k]).  M must be multiple of 16.
// Block = 128 threads (4 waves); each wave owns a 16x64 output strip.
__global__ __launch_bounds__(128)
void k_gemm_wmma(const _Float16* __restrict__ A,
                 const _Float16* __restrict__ Bt,
                 const float* __restrict__ bias,
                 float* __restrict__ Cout) {
    const int lane = threadIdx.x & 31;
    const int wave = threadIdx.x >> 5;
    const int grp  = lane >> 4;        // 0 or 1 (half-wave group)
    const int l16  = lane & 15;

    const int m0     = blockIdx.y * 16;                 // row tile base
    const int ntBase = (blockIdx.x * 4 + wave) * 64;    // 64-wide column strip

    f8 acc[4];
#pragma unroll
    for (int j = 0; j < 4; ++j) {
        float bv = bias[ntBase + j * 16 + l16];
#pragma unroll
        for (int v = 0; v < 8; ++v) acc[j][v] = bv;
    }

    const _Float16* aRow = A + (size_t)(m0 + l16) * C;  // row m = lane&15
    const _Float16* bRow[4];
#pragma unroll
    for (int j = 0; j < 4; ++j)
        bRow[j] = Bt + (size_t)(ntBase + j * 16 + l16) * C;  // col n (transposed)

    // explicit double buffering: set 0 and set 1 alternate every 32 of K.
    // sched_barrier(0) after each load batch pins the issue order:
    //   loads(next set) must stay ABOVE the wmma group consuming the other set,
    // forcing distinct registers + partial s_wait_loadcnt (loads in flight
    // while the matrix pipe works).
    Frag a0, a1, b0[4], b1[4];
    load_a(a0, aRow, 0, grp);
#pragma unroll
    for (int j = 0; j < 4; ++j) load_b(b0[j], bRow[j], 0, grp);
    __builtin_amdgcn_sched_barrier(0);

#pragma unroll
    for (int kk = 0; kk < C; kk += 64) {
        // issue loads for the second half-step before consuming the first
        load_a(a1, aRow, kk + 32, grp);
#pragma unroll
        for (int j = 0; j < 4; ++j) load_b(b1[j], bRow[j], kk + 32, grp);
        if (kk + 64 < C)   // prefetch next A chunk (global_prefetch_b8)
            __builtin_prefetch(aRow + kk + 64, 0, 3);
        __builtin_amdgcn_sched_barrier(0);

#pragma unroll
        for (int j = 0; j < 4; ++j)
            acc[j] = __builtin_amdgcn_wmma_f32_16x16x32_f16(
                         false, a0.v, false, b0[j].v, (short)0, acc[j], false, false);

        if (kk + 64 < C) {  // issue loads for next iteration's first half-step
            load_a(a0, aRow, kk + 64, grp);
#pragma unroll
            for (int j = 0; j < 4; ++j) load_b(b0[j], bRow[j], kk + 64, grp);
        }
        __builtin_amdgcn_sched_barrier(0);

#pragma unroll
        for (int j = 0; j < 4; ++j)
            acc[j] = __builtin_amdgcn_wmma_f32_16x16x32_f16(
                         false, a1.v, false, b1[j].v, (short)0, acc[j], false, false);
        __builtin_amdgcn_sched_barrier(0);
    }

#pragma unroll
    for (int j = 0; j < 4; ++j) {
        float* cPtr = Cout + (size_t)(m0 + grp * 8) * C + ntBase + j * 16 + l16;
#pragma unroll
        for (int v = 0; v < 8; ++v) cPtr[(size_t)v * C] = acc[j][v];
    }
}

// ---------------- graph aggregation --------------------------------------
// self-loop contribution also fully initializes the output buffer
__global__ void k_agg_init(const float* __restrict__ H, const float* __restrict__ dinv,
                           float* __restrict__ out) {
    int i = blockIdx.x;
    float w = dinv[i] * dinv[i];
    const float* h = H + (size_t)i * C;
    float* o = out + (size_t)i * C;
    for (int c = threadIdx.x; c < C; c += blockDim.x) o[c] = w * h[c];
}

// out[src] += dinv[src]*dinv[dst] * H[dst]
__global__ void k_agg_edges(const float* __restrict__ H, const float* __restrict__ dinv,
                            const int* __restrict__ src, const int* __restrict__ dst,
                            float* __restrict__ out) {
    int e = blockIdx.x;
    int s = src[e], d = dst[e];
    float w = dinv[s] * dinv[d];
    const float* h = H + (size_t)d * C;
    float* o = out + (size_t)s * C;
    for (int c = threadIdx.x; c < C; c += blockDim.x)
        atomicAdd(&o[c], w * h[c]);
}

// ---------------- batch norm ---------------------------------------------
__global__ __launch_bounds__(256)
void k_bn_stats(const float* __restrict__ H, const float* __restrict__ gamma,
                const float* __restrict__ beta, float* __restrict__ scale,
                float* __restrict__ shift, int N) {
    int c = blockIdx.x;                 // one block per channel
    float s = 0.0f, sq = 0.0f;
    for (int i = threadIdx.x; i < N; i += 256) {
        float v = H[(size_t)i * C + c];
        s += v; sq += v * v;
    }
    __shared__ float ls[256], lq[256];
    ls[threadIdx.x] = s; lq[threadIdx.x] = sq;
    __syncthreads();
    for (int st = 128; st > 0; st >>= 1) {
        if (threadIdx.x < st) {
            ls[threadIdx.x] += ls[threadIdx.x + st];
            lq[threadIdx.x] += lq[threadIdx.x + st];
        }
        __syncthreads();
    }
    if (threadIdx.x == 0) {
        float inv_n = 1.0f / (float)N;
        float mu  = ls[0] * inv_n;
        float var = lq[0] * inv_n - mu * mu;          // biased variance
        float r   = rsqrtf(var + BN_EPS);
        float sc  = gamma[c] * r;
        scale[c] = sc;
        shift[c] = beta[c] - mu * sc;
    }
}

// fused BN apply + ReLU + f16 convert (feeds GEMM2)
__global__ void k_bn_relu_cvt(const float* __restrict__ H, const float* __restrict__ scale,
                              const float* __restrict__ shift, _Float16* __restrict__ out,
                              int total) {
    int i = blockIdx.x * blockDim.x + threadIdx.x;
    if (i < total) {
        int c = i & (C - 1);
        float v = fmaf(H[i], scale[c], shift[c]);
        out[i] = (_Float16)fmaxf(v, 0.0f);
    }
}

// ---------------------------------------------------------------------------
extern "C" void kernel_launch(void* const* d_in, const int* in_sizes, int n_in,
                              void* d_out, int out_size, void* d_ws, size_t ws_size,
                              hipStream_t stream) {
    const float* x     = (const float*)d_in[0];
    const float* W1    = (const float*)d_in[1];
    const float* b1    = (const float*)d_in[2];
    const float* gamma = (const float*)d_in[3];
    const float* beta  = (const float*)d_in[4];
    const float* W2    = (const float*)d_in[5];
    const float* b2    = (const float*)d_in[6];
    const int*   ei    = (const int*)d_in[7];   // [2,E] flattened: src then dst

    const int N = in_sizes[0] / C;
    const int E = in_sizes[7] / 2;
    const int* e_src = ei;
    const int* e_dst = ei + E;
    float* out = (float*)d_out;

    // ---- workspace carve-out (256B aligned) ----
    char*  w   = (char*)d_ws;
    size_t off = 0;
    auto carve = [&](size_t bytes) -> void* {
        void* p = w + off;
        off = (off + bytes + 255) & ~(size_t)255;
        return p;
    };
    float*     dinv  = (float*)    carve((size_t)N * 4);
    float*     scale = (float*)    carve((size_t)C * 4);
    float*     shift = (float*)    carve((size_t)C * 4);
    _Float16*  xh    = (_Float16*) carve((size_t)N * C * 2);
    _Float16*  W1t   = (_Float16*) carve((size_t)C * C * 2);
    _Float16*  W2t   = (_Float16*) carve((size_t)C * C * 2);
    float*     Hg    = (float*)    carve((size_t)N * C * 4);  // GEMM outputs (reused)
    float*     Ha    = (float*)    carve((size_t)N * C * 4);  // agg-1 output
    _Float16*  H2h   = (_Float16*) carve((size_t)N * C * 2);  // BN+ReLU f16
    (void)ws_size; (void)n_in; (void)out_size;

    const int total = N * C;
    const dim3 gemm_grid(C / 256, N / 16);  // 4 waves/block, each a 16x64 strip

    // 1) symmetric normalization weights
    k_zero_f32 <<<(N + 255) / 256, 256, 0, stream>>>(dinv, N);
    k_deg_count<<<(E + 255) / 256, 256, 0, stream>>>(e_dst, dinv, E);
    k_dinv     <<<(N + 255) / 256, 256, 0, stream>>>(dinv, N);

    // 2) f16 operands (weights transposed for contiguous B fragments)
    k_cvt_f16      <<<(total + 255) / 256, 256, 0, stream>>>(x, xh, total);
    k_cvt_transpose<<<C, 256, 0, stream>>>(W1, W1t);
    k_cvt_transpose<<<C, 256, 0, stream>>>(W2, W2t);

    // 3) layer 1: GEMM -> agg -> BN+ReLU
    k_gemm_wmma<<<gemm_grid, 128, 0, stream>>>(xh, W1t, b1, Hg);
    k_agg_init <<<N, 256, 0, stream>>>(Hg, dinv, Ha);
    k_agg_edges<<<E, 256, 0, stream>>>(Hg, dinv, e_src, e_dst, Ha);
    k_bn_stats <<<C, 256, 0, stream>>>(Ha, gamma, beta, scale, shift, N);
    k_bn_relu_cvt<<<(total + 255) / 256, 256, 0, stream>>>(Ha, scale, shift, H2h, total);

    // 4) layer 2: GEMM -> agg -> d_out
    k_gemm_wmma<<<gemm_grid, 128, 0, stream>>>(H2h, W2t, b2, Hg);
    k_agg_init <<<N, 256, 0, stream>>>(Hg, dinv, out);
    k_agg_edges<<<E, 256, 0, stream>>>(Hg, dinv, e_src, e_dst, out);
}
